// MyLLMFFNCoE_78718160601091
// MI455X (gfx1250) — compile-verified, matrix operand order checked by reading, not run on hardware
//
#include <hip/hip_runtime.h>
#include <hip/hip_bf16.h>

// ---------------------------------------------------------------------------
// MyLLMFFNCoE on MI455X (gfx1250).
// GEMMs: v_wmma_f32_16x16x32_bf16, bf16 operands staged to LDS with
// global_load_async_to_lds_b128 (double buffered), fp32 accumulation.
// Weights converted+transposed to bf16 once per launch; activations kept bf16.
// ---------------------------------------------------------------------------

typedef __attribute__((ext_vector_type(16))) __bf16 v16bf;
typedef __attribute__((ext_vector_type(8)))  float  v8f;

#define HDIM 1024
#define IDIM 1024
#define NEXP 6
#define NTOK 8192
#define SEG  2048
#define NCH  2
#define EPSV 1e-6f

// GEMM tiling: 128x128 block tile, 4 waves, each wave 64x64 (4x4 WMMA tiles).
#define BM 128
#define BN 128
#define BK 64
#define ROWB 144                    // LDS row pitch bytes: 64*2 + 16 pad (16B aligned)
#define TILEB (BM * ROWB)           // one A or B tile: 18432 B
#define BUFB  (2 * TILEB)           // A + B tile: 36864 B
// double buffered: 73728 B static LDS

__device__ __forceinline__ void async_copy_b128(unsigned lds_off, const void* gaddr) {
    asm volatile("global_load_async_to_lds_b128 %0, %1, off"
                 :: "v"(lds_off), "v"(gaddr) : "memory");
}
__device__ __forceinline__ void wait_asynccnt0() {
    asm volatile("s_wait_asynccnt 0x0" ::: "memory");
}

// Fragment: 16 bf16/lane. ISA 16-bit layout: lanes 0-15 K 0..7 & 16..23,
// lanes 16-31 K 8..15 & 24..31; row/col = lane & 15.
__device__ __forceinline__ v16bf load_frag(const char* rowp, int lane) {
    const int kl = (lane >> 4) * 8;
    const __bf16* p = (const __bf16*)rowp;
    v16bf f;
#pragma unroll
    for (int v = 0; v < 8; ++v) {
        const int k = kl + (v & 3) * 2 + ((v >> 2) << 4);
        f[2 * v]     = p[k];
        f[2 * v + 1] = p[k + 1];
    }
    return f;
}

// MODE 0: OUT(bf16) = A@B + bias
// MODE 1: OUT(bf16) = silu(A@B + bias) * hprev
// MODE 2: ACC(f32) += scale_row * (A@B + bias)
template <int MODE>
__global__ __launch_bounds__(128) void gemm_kernel(
    const __bf16* __restrict__ A, int lda,        // activations, row-major MxK
    const __bf16* __restrict__ BT, int ldb,       // weights transposed, NxK
    const float* __restrict__ bias,               // per output column
    const __bf16* __restrict__ hprev,             // MODE 1 (ld = IDIM)
    const float* __restrict__ scale,              // MODE 2: per-row, stride NEXP; null => 1
    __bf16* __restrict__ obf,                     // MODE 0/1 output
    float* __restrict__ oacc,                     // MODE 2 accumulator
    int ldo, int K)
{
    __shared__ __align__(16) char smem[2 * BUFB];

    const int tid   = threadIdx.x;
    const int lane  = tid & 31;
    const int wave  = tid >> 5;          // 0..3
    const int waveM = wave >> 1;         // 0..1 -> 64 rows
    const int waveN = wave & 1;          // 0..1 -> 64 cols
    const int bm0   = blockIdx.y * BM;
    const int bn0   = blockIdx.x * BN;

    const v8f vzero = {0.f, 0.f, 0.f, 0.f, 0.f, 0.f, 0.f, 0.f};
    v8f acc[4][4];
#pragma unroll
    for (int mi = 0; mi < 4; ++mi)
#pragma unroll
        for (int ni = 0; ni < 4; ++ni) acc[mi][ni] = vzero;

    const unsigned smem_base = (unsigned)(size_t)(void*)&smem[0];
    const int NKT = K / BK;

    // ---- stage one K-slab (A + BT tiles) into buffer b via async copies ----
    auto stage = [&](int b, int kt) {
        const int k0 = kt * BK;
        const unsigned abase = smem_base + (unsigned)b * BUFB;
        const unsigned bbase = abase + TILEB;
        const char* Ab = (const char*)A  + ((size_t)bm0 * lda + k0) * 2;
        const char* Bb = (const char*)BT + ((size_t)bn0 * ldb + k0) * 2;
#pragma unroll
        for (int i = 0; i < 8; ++i) {                 // 1024 chunks / 128 thr
            const int idx = tid + 128 * i;
            const int row = idx >> 3;                 // 8 x 16B chunks per row
            const int c   = idx & 7;
            async_copy_b128(abase + row * ROWB + c * 16,
                            Ab + (size_t)row * lda * 2 + c * 16);
        }
#pragma unroll
        for (int i = 0; i < 8; ++i) {
            const int idx = tid + 128 * i;
            const int row = idx >> 3;
            const int c   = idx & 7;
            async_copy_b128(bbase + row * ROWB + c * 16,
                            Bb + (size_t)row * ldb * 2 + c * 16);
        }
    };

    stage(0, 0);
    const int rsel = lane & 15;
    for (int kt = 0; kt < NKT; ++kt) {
        wait_asynccnt0();
        __syncthreads();
        if (kt + 1 < NKT) stage((kt + 1) & 1, kt + 1);

        const char* sA = smem + (kt & 1) * BUFB;
        const char* sB = sA + TILEB;
#pragma unroll
        for (int ks = 0; ks < BK; ks += 32) {
            v16bf afr[4], bfr[4];
#pragma unroll
            for (int mi = 0; mi < 4; ++mi)
                afr[mi] = load_frag(sA + (waveM * 64 + mi * 16 + rsel) * ROWB + ks * 2, lane);
#pragma unroll
            for (int ni = 0; ni < 4; ++ni)
                bfr[ni] = load_frag(sB + (waveN * 64 + ni * 16 + rsel) * ROWB + ks * 2, lane);
#pragma unroll
            for (int mi = 0; mi < 4; ++mi)
#pragma unroll
                for (int ni = 0; ni < 4; ++ni)
                    acc[mi][ni] = __builtin_amdgcn_wmma_f32_16x16x32_bf16(
                        false, afr[mi], false, bfr[ni], (short)0, acc[mi][ni], false, false);
        }
        __syncthreads();
    }

    // ---- epilogue; C/D layout: n = lane&15, m = (lane>>4)*8 + v ----
    const int cl    = lane & 15;
    const int rhalf = (lane >> 4) * 8;
#pragma unroll
    for (int mi = 0; mi < 4; ++mi) {
#pragma unroll
        for (int ni = 0; ni < 4; ++ni) {
            const int col = bn0 + waveN * 64 + ni * 16 + cl;
            const float bv = bias[col];
#pragma unroll
            for (int v = 0; v < 8; ++v) {
                const int row = bm0 + waveM * 64 + mi * 16 + rhalf + v;
                const float val = acc[mi][ni][v] + bv;
                if (MODE == 0) {
                    obf[(size_t)row * ldo + col] = (__bf16)val;
                } else if (MODE == 1) {
                    const float sg = val / (1.f + __expf(-val));
                    obf[(size_t)row * ldo + col] =
                        (__bf16)(sg * (float)hprev[(size_t)row * IDIM + col]);
                } else {
                    const float sc = scale ? scale[(size_t)row * NEXP] : 1.0f;
                    float* p = &oacc[(size_t)row * ldo + col];
                    *p += sc * val;
                }
            }
        }
    }
}

// ---- weight convert + transpose: W[K][N] f32 -> WT[N][K] bf16 (batched z) ----
__global__ __launch_bounds__(256) void convT_kernel(
    const float* __restrict__ W, __bf16* __restrict__ WT, int K, int N)
{
    __shared__ float tile[32][33];
    const size_t zoff = (size_t)blockIdx.z * K * N;
    const int bn = blockIdx.x * 32;
    const int bk = blockIdx.y * 32;
    const int tx = threadIdx.x & 31;
    const int ty = threadIdx.x >> 5;     // 0..7
#pragma unroll
    for (int i = ty; i < 32; i += 8)
        tile[i][tx] = W[zoff + (size_t)(bk + i) * N + bn + tx];
    __syncthreads();
#pragma unroll
    for (int i = ty; i < 32; i += 8)
        WT[zoff + (size_t)(bn + i) * K + bk + tx] = (__bf16)tile[tx][i];
}

// ---- init: acc = x (f32 residual), xbf = bf16(x) ----
__global__ void init_kernel(const float* __restrict__ x, float* __restrict__ acc,
                            __bf16* __restrict__ xbf, int n4)
{
    const int i = blockIdx.x * blockDim.x + threadIdx.x;
    if (i >= n4) return;
    const float4 v = ((const float4*)x)[i];
    ((float4*)acc)[i] = v;
    __bf16* o = &xbf[(size_t)i * 4];
    o[0] = (__bf16)v.x; o[1] = (__bf16)v.y; o[2] = (__bf16)v.z; o[3] = (__bf16)v.w;
}

// ---- router: top-2 of 6 biased logits, softmax over raw logits ----
__global__ __launch_bounds__(256) void router_kernel(
    const float* __restrict__ xf, const float* __restrict__ rW,
    const float* __restrict__ rb, const float* __restrict__ gb,
    float* __restrict__ w, int ntok)
{
    const int lane = threadIdx.x & 31;
    const int t    = blockIdx.x * 8 + (threadIdx.x >> 5);
    if (t >= ntok) return;
    const float* xp = xf + (size_t)t * HDIM;
    float g[NEXP];
#pragma unroll
    for (int e = 0; e < NEXP; ++e) g[e] = 0.f;
    for (int h = lane; h < HDIM; h += 32) {
        const float xv = xp[h];
#pragma unroll
        for (int e = 0; e < NEXP; ++e) g[e] += xv * rW[h * NEXP + e];
    }
#pragma unroll
    for (int e = 0; e < NEXP; ++e)
#pragma unroll
        for (int off = 16; off > 0; off >>= 1) g[e] += __shfl_xor(g[e], off, 32);

    if (lane == 0) {
        float gate[NEXP], biased[NEXP];
#pragma unroll
        for (int e = 0; e < NEXP; ++e) {
            gate[e]   = g[e] + rb[e];           // TEMP == 1
            biased[e] = gate[e] + gb[e];
        }
        int i1 = 0;
#pragma unroll
        for (int e = 1; e < NEXP; ++e) if (biased[e] > biased[i1]) i1 = e;
        int i2 = (i1 == 0) ? 1 : 0;
#pragma unroll
        for (int e = 0; e < NEXP; ++e)
            if (e != i1 && biased[e] > biased[i2]) i2 = e;
        const float m  = fmaxf(gate[i1], gate[i2]);
        const float e1 = __expf(gate[i1] - m), e2 = __expf(gate[i2] - m);
        const float inv = 1.f / (e1 + e2);
#pragma unroll
        for (int e = 0; e < NEXP; ++e) w[(size_t)t * NEXP + e] = 0.f;
        w[(size_t)t * NEXP + i1] = e1 * inv;
        w[(size_t)t * NEXP + i2] = e2 * inv;
    }
}

// ---- RMSNorm finalize: dstf = y*rsqrt(mean(y^2)+eps)*nw ; xbf = bf16(dstf) ----
__global__ __launch_bounds__(256) void finalize_kernel(
    const float* __restrict__ accp, const float* __restrict__ nw,
    float* __restrict__ dstf, __bf16* __restrict__ xbf, int ntok)
{
    const int lane = threadIdx.x & 31;
    const int t    = blockIdx.x * 8 + (threadIdx.x >> 5);
    if (t >= ntok) return;
    const float* yp = accp + (size_t)t * HDIM;
    float s = 0.f;
    for (int h = lane; h < HDIM; h += 32) { const float y = yp[h]; s += y * y; }
#pragma unroll
    for (int off = 16; off > 0; off >>= 1) s += __shfl_xor(s, off, 32);
    const float r = rsqrtf(s * (1.0f / HDIM) + EPSV);
    for (int h = lane; h < HDIM; h += 32) {
        const float y = yp[h] * r * nw[h];
        dstf[(size_t)t * HDIM + h] = y;
        xbf[(size_t)t * HDIM + h]  = (__bf16)y;
    }
}

extern "C" void kernel_launch(void* const* d_in, const int* in_sizes, int n_in,
                              void* d_out, int out_size, void* d_ws, size_t ws_size,
                              hipStream_t stream) {
    const float* x    = (const float*)d_in[0];
    const float* rW   = (const float*)d_in[1];   // (C, H, E)
    const float* rb   = (const float*)d_in[2];   // (C, E)
    const float* gb   = (const float*)d_in[3];   // (C, E)
    const float* eWup = (const float*)d_in[4];   // (E, H, I)
    const float* eBup = (const float*)d_in[5];
    const float* eWg  = (const float*)d_in[6];   // (E, I, I)
    const float* eBg  = (const float*)d_in[7];
    const float* eWdn = (const float*)d_in[8];   // (E, I, H)
    const float* eBdn = (const float*)d_in[9];
    const float* sWup = (const float*)d_in[10];  // (H, 2I)
    const float* sBup = (const float*)d_in[11];
    const float* sWg  = (const float*)d_in[12];  // (I, I)
    const float* sBg  = (const float*)d_in[13];
    const float* sWdn = (const float*)d_in[14];  // (I, H)
    const float* sBdn = (const float*)d_in[15];
    const float* nrm  = (const float*)d_in[16];  // (C, H)

    // ---- workspace layout ----
    char* ws = (char*)d_ws;
    size_t off = 0;
    float*  accb  = (float*)(ws + off);  off += (size_t)NTOK * HDIM * 4;     // 33.5M
    __bf16* xbf   = (__bf16*)(ws + off); off += (size_t)NTOK * HDIM * 2;     // 16.8M
    __bf16* hbuf  = (__bf16*)(ws + off); off += (size_t)SEG * IDIM * 2;      // 4.2M
    __bf16* gbuf  = (__bf16*)(ws + off); off += (size_t)SEG * IDIM * 2;      // 4.2M
    float*  wbuf  = (float*)(ws + off);  off += (size_t)NTOK * NEXP * 4;     // 0.2M
    __bf16* eWupT = (__bf16*)(ws + off); off += (size_t)NEXP * HDIM * IDIM * 2;
    __bf16* eWgT  = (__bf16*)(ws + off); off += (size_t)NEXP * IDIM * IDIM * 2;
    __bf16* eWdnT = (__bf16*)(ws + off); off += (size_t)NEXP * IDIM * HDIM * 2;
    __bf16* sWupT = (__bf16*)(ws + off); off += (size_t)HDIM * 2 * IDIM * 2;
    __bf16* sWgT  = (__bf16*)(ws + off); off += (size_t)IDIM * IDIM * 2;
    __bf16* sWdnT = (__bf16*)(ws + off); off += (size_t)IDIM * HDIM * 2;     // total ~105MB

    // ---- one-time weight convert + transpose ----
    convT_kernel<<<dim3(IDIM / 32, HDIM / 32, NEXP), 256, 0, stream>>>(eWup, eWupT, HDIM, IDIM);
    convT_kernel<<<dim3(IDIM / 32, IDIM / 32, NEXP), 256, 0, stream>>>(eWg,  eWgT,  IDIM, IDIM);
    convT_kernel<<<dim3(HDIM / 32, IDIM / 32, NEXP), 256, 0, stream>>>(eWdn, eWdnT, IDIM, HDIM);
    convT_kernel<<<dim3(2 * IDIM / 32, HDIM / 32, 1), 256, 0, stream>>>(sWup, sWupT, HDIM, 2 * IDIM);
    convT_kernel<<<dim3(IDIM / 32, IDIM / 32, 1), 256, 0, stream>>>(sWg, sWgT, IDIM, IDIM);
    convT_kernel<<<dim3(HDIM / 32, IDIM / 32, 1), 256, 0, stream>>>(sWdn, sWdnT, IDIM, HDIM);

    init_kernel<<<(NTOK * HDIM / 4 + 255) / 256, 256, 0, stream>>>(
        x, accb, xbf, NTOK * HDIM / 4);

    const dim3 gblk(128);
    const dim3 ggrid(HDIM / BN, SEG / BM);   // (8, 16)

    for (int j = 0; j < NCH; ++j) {
        for (int sgi = 0; sgi < NTOK / SEG; ++sgi) {
            const int tok0 = sgi * SEG;
            const __bf16* xseg = xbf + (size_t)tok0 * HDIM;
            float* aseg = accb + (size_t)tok0 * HDIM;
            float* wseg = wbuf + (size_t)tok0 * NEXP;

            // router reads the f32 residual (== xf before experts touch acc)
            router_kernel<<<SEG / 8, 256, 0, stream>>>(
                aseg, rW + (size_t)j * HDIM * NEXP, rb + j * NEXP, gb + j * NEXP, wseg, SEG);

            for (int e = 0; e < NEXP; ++e) {
                gemm_kernel<0><<<ggrid, gblk, 0, stream>>>(
                    xseg, HDIM, eWupT + (size_t)e * HDIM * IDIM, HDIM,
                    eBup + (size_t)e * IDIM, nullptr, nullptr, hbuf, nullptr, IDIM, HDIM);
                gemm_kernel<1><<<ggrid, gblk, 0, stream>>>(
                    hbuf, IDIM, eWgT + (size_t)e * IDIM * IDIM, IDIM,
                    eBg + (size_t)e * IDIM, hbuf, nullptr, gbuf, nullptr, IDIM, IDIM);
                gemm_kernel<2><<<ggrid, gblk, 0, stream>>>(
                    gbuf, IDIM, eWdnT + (size_t)e * IDIM * HDIM, IDIM,
                    eBdn + (size_t)e * HDIM, nullptr, wseg + e, nullptr, aseg, HDIM, IDIM);
            }
            for (int k = 0; k < 2; ++k) {    // NS share experts
                gemm_kernel<0><<<ggrid, gblk, 0, stream>>>(
                    xseg, HDIM, sWupT + (size_t)k * IDIM * HDIM, HDIM,
                    sBup + (size_t)k * IDIM, nullptr, nullptr, hbuf, nullptr, IDIM, HDIM);
                gemm_kernel<1><<<ggrid, gblk, 0, stream>>>(
                    hbuf, IDIM, sWgT, IDIM, sBg, hbuf, nullptr, gbuf, nullptr, IDIM, IDIM);
                gemm_kernel<2><<<ggrid, gblk, 0, stream>>>(
                    gbuf, IDIM, sWdnT, IDIM, sBdn, nullptr, nullptr, nullptr, aseg, HDIM, IDIM);
            }
        }
        float* dstf = (j == NCH - 1) ? (float*)d_out : accb;
        finalize_kernel<<<NTOK / 8, 256, 0, stream>>>(
            accb, nrm + (size_t)j * HDIM, dstf, xbf, NTOK);
    }
}
